// Net_33724083208203
// MI455X (gfx1250) — matrix operand order repaired and checked by reading, not compiled, not run on hardware
//
#include <hip/hip_runtime.h>
#include <hip/hip_bf16.h>

// ---------------------------------------------------------------------------
// Graph-transformer energy gradient for MI455X (gfx1250, wave32, WMMA).
// - Edge-feature tensor (B,N,N,INNER) never materialized: rank-3 corrections.
// - All GEMMs: v_wmma_f32_16x16x32_f16, f16 in / f32 acc, single contiguous
//   (A row-major, B K-contiguous) form; operands that would need strided
//   access get tiny f16 transposed copies instead.
// - Fragment loads are b128-vectorized; epilogue is branch-free via templates.
// ---------------------------------------------------------------------------

typedef _Float16 half_t;
typedef __attribute__((ext_vector_type(16))) _Float16 v16h;
typedef __attribute__((ext_vector_type(8)))  _Float16 v8h;
typedef __attribute__((ext_vector_type(8)))  float    v8f;

#define Bsz   2
#define Nn    128
#define Dm    64
#define Hh    8
#define DHd   64
#define INNER 512
#define FFD   256
#define En    256
#define NLAY  6
#define APPS  12
#define SCALE 0.125f
#define ROWS  (Bsz*Nn)     // 256

__device__ __forceinline__ float wred_sum(float v) {
  #pragma unroll
  for (int m = 16; m; m >>= 1) v += __shfl_xor(v, m, 32);
  return v;
}
__device__ __forceinline__ float wred_max(float v) {
  #pragma unroll
  for (int m = 16; m; m >>= 1) v = fmaxf(v, __shfl_xor(v, m, 32));
  return v;
}

// ------------------------------ utility kernels ----------------------------

__global__ void k_zero(float* p, int n) {
  int i = blockIdx.x * blockDim.x + threadIdx.x;
  for (; i < n; i += gridDim.x * blockDim.x) p[i] = 0.f;
}

__global__ void k_cvt(const float* __restrict__ s, half_t* __restrict__ d, int n) {
  int i = blockIdx.x * blockDim.x + threadIdx.x;
  for (; i < n; i += gridDim.x * blockDim.x) d[i] = (half_t)s[i];
}

// dst[c*R + r] = src[r*C + c]
__global__ void k_cvt_T(const float* __restrict__ s, half_t* __restrict__ d, int R, int C) {
  int i = blockIdx.x * blockDim.x + threadIdx.x;
  int n = R * C;
  for (; i < n; i += gridDim.x * blockDim.x) {
    int r = i / C, c = i % C;
    d[c * R + r] = (half_t)s[i];
  }
}

// Per-head transpose: dst[(b*H+h)*DHd*Nn + d*Nn + j] = src[(b*N+j)*ld + off + h*DHd + d]
__global__ void k_tr_head(const float* __restrict__ src, half_t* __restrict__ dst,
                          int ld, int off) {
  int i = blockIdx.x * blockDim.x + threadIdx.x;        // over B*H*DHd*Nn
  if (i >= Bsz * Hh * DHd * Nn) return;
  int bh = i >> 13, r = i & 8191, d = r >> 7, j = r & 127;
  int b = bh >> 3, h = bh & 7;
  dst[i] = (half_t)src[((long)(b * Nn + j)) * ld + off + h * DHd + d];
}

__global__ void k_build_nodes(const int* __restrict__ idx, const float* __restrict__ emb,
                              const float* __restrict__ noise, float* __restrict__ nodes) {
  int i = blockIdx.x * blockDim.x + threadIdx.x;
  if (i >= ROWS * Dm) return;
  int d = i % Dm, bn = i / Dm, b = bn / Nn;
  nodes[i] = (d < 63) ? emb[idx[bn] * 63 + d] : noise[b];
}

__global__ void k_build_edges(const int* __restrict__ bonds, const float* __restrict__ coords,
                              float* __restrict__ edges, int* __restrict__ mask) {
  int e = blockIdx.x * blockDim.x + threadIdx.x;
  if (e >= En) return;
  int i = bonds[e * 2 + 0], j = bonds[e * 2 + 1];
  for (int b = 0; b < Bsz; ++b)
    for (int c = 0; c < 3; ++c) {
      float diff = coords[(b * Nn + i) * 3 + c] - coords[(b * Nn + j) * 3 + c];
      edges[(((long)b * Nn + i) * Nn + j) * 3 + c] = diff;
      edges[(((long)b * Nn + j) * Nn + i) * 3 + c] = -diff;
    }
  mask[i * Nn + j] = 1;
  mask[j * Nn + i] = 1;
}

// -------------------------------- layernorm --------------------------------

__global__ void k_ln_fwd(const float* __restrict__ x, const float* __restrict__ g,
                         const float* __restrict__ bt, half_t* __restrict__ yh,
                         float* __restrict__ mean, float* __restrict__ rstd) {
  int row = blockIdx.x, t = threadIdx.x;
  float v0 = x[row * Dm + t], v1 = x[row * Dm + t + 32];
  float m = wred_sum(v0 + v1) * (1.f / Dm);
  float sq = wred_sum(v0 * v0 + v1 * v1) * (1.f / Dm);
  float rs = rsqrtf(sq - m * m + 1e-5f);
  yh[row * Dm + t]      = (half_t)((v0 - m) * rs * g[t] + bt[t]);
  yh[row * Dm + t + 32] = (half_t)((v1 - m) * rs * g[t + 32] + bt[t + 32]);
  if (t == 0) { mean[row] = m; rstd[row] = rs; }
}

__global__ void k_ln_bwd(const float* __restrict__ dy, const float* __restrict__ xin,
                         const float* __restrict__ g, const float* __restrict__ mean,
                         const float* __restrict__ rstd, float* __restrict__ dn, int acc) {
  int row = blockIdx.x, t = threadIdx.x;
  float m = mean[row], rs = rstd[row];
  float x0 = xin[row * Dm + t], x1 = xin[row * Dm + t + 32];
  float xh0 = (x0 - m) * rs, xh1 = (x1 - m) * rs;
  float a0 = g[t] * dy[row * Dm + t], a1 = g[t + 32] * dy[row * Dm + t + 32];
  float s1 = wred_sum(a0 + a1) * (1.f / Dm);
  float s2 = wred_sum(a0 * xh0 + a1 * xh1) * (1.f / Dm);
  float d0 = rs * (a0 - s1 - xh0 * s2), d1 = rs * (a1 - s1 - xh1 * s2);
  if (acc) { dn[row * Dm + t] += d0; dn[row * Dm + t + 32] += d1; }
  else     { dn[row * Dm + t]  = d0; dn[row * Dm + t + 32]  = d1; }
}

// ------------------------------ WMMA GEMM ----------------------------------
// C[M,N] = alpha * A[M,K] @ Bt[N,K]^T (+bias) (+C) ; optional f16 mirror.
// A row-major (lda >= K-run contiguous); Bt row-major over N with K contiguous.
// All K-runs are 8-half (16B) aligned -> b128 vector loads.

template <int BIAS, int BETA, int CH>
__global__ void gemm_nt(const half_t* __restrict__ A, const half_t* __restrict__ B,
                        const float* __restrict__ bias, float* __restrict__ C,
                        half_t* __restrict__ Ch, int M, int N, int K,
                        int lda, int ldb, int ldc,
                        long sAb, long sAh, long sBb, long sBh, long sCb, long sCh,
                        int Hdiv, float alpha) {
  int z = blockIdx.z, bb = z / Hdiv, hh = z % Hdiv;
  A += (long)bb * sAb + (long)hh * sAh;
  B += (long)bb * sBb + (long)hh * sBh;
  long coff = (long)bb * sCb + (long)hh * sCh;
  C += coff; if (CH) Ch += coff;

  int m0 = blockIdx.x * 16;
  int n0 = blockIdx.y * 32 + threadIdx.y * 16;
  int lane = threadIdx.x;
  int r = lane & 15;
  int koffA = (lane < 16) ? 0 : 8;
  int koffB = (lane < 16) ? 0 : 8;
  const half_t* Arow = A + (long)(m0 + r) * lda;
  const half_t* Brow = B + (long)(n0 + r) * ldb;

  v8f acc = {};
  for (int k0 = 0; k0 < K; k0 += 32) {
    v8h al = *(const v8h*)(Arow + k0 + koffA);
    v8h ah = *(const v8h*)(Arow + k0 + 16 + koffA);
    v8h bl = *(const v8h*)(Brow + k0 + koffB);
    v8h bh = *(const v8h*)(Brow + k0 + 16 + koffB);
    v16h a = __builtin_shufflevector(al, ah, 0, 1, 2, 3, 4, 5, 6, 7,
                                     8, 9, 10, 11, 12, 13, 14, 15);
    v16h b = __builtin_shufflevector(bl, bh, 0, 1, 2, 3, 4, 5, 6, 7,
                                     8, 9, 10, 11, 12, 13, 14, 15);
    acc = __builtin_amdgcn_wmma_f32_16x16x32_f16(false, a, false, b, (short)0, acc,
                                                 false, false);
  }
  int col = n0 + r;
  int rbase = (lane < 16) ? 0 : 8;
  float bv = BIAS ? bias[col] : 0.f;
  #pragma unroll
  for (int t = 0; t < 8; ++t) {
    long idx = (long)(m0 + rbase + t) * ldc + col;
    float v = acc[t] * alpha;
    if (BIAS) v += bv;
    if (BETA) v += C[idx];
    C[idx] = v;
    if (CH) Ch[idx] = (half_t)v;
  }
}

static void gemm_launch(hipStream_t s, int BIAS, int BETA, int CHf,
                        const half_t* A, const half_t* B, const float* bias,
                        float* C, half_t* Ch, int M, int N, int K,
                        int lda, int ldb, int ldc, long sAb, long sAh, long sBb, long sBh,
                        long sCb, long sCh, int Hdiv, int batches, float alpha) {
  dim3 g(M / 16, N / 32, batches), b(32, 2, 1);
#define GA A, B, bias, C, Ch, M, N, K, lda, ldb, ldc, sAb, sAh, sBb, sBh, sCb, sCh, Hdiv, alpha
  int sel = BIAS * 4 + BETA * 2 + CHf;
  switch (sel) {
    case 0: gemm_nt<0, 0, 0><<<g, b, 0, s>>>(GA); break;
    case 1: gemm_nt<0, 0, 1><<<g, b, 0, s>>>(GA); break;
    case 2: gemm_nt<0, 1, 0><<<g, b, 0, s>>>(GA); break;
    case 3: gemm_nt<0, 1, 1><<<g, b, 0, s>>>(GA); break;
    case 4: gemm_nt<1, 0, 0><<<g, b, 0, s>>>(GA); break;
    case 5: gemm_nt<1, 0, 1><<<g, b, 0, s>>>(GA); break;
    case 6: gemm_nt<1, 1, 0><<<g, b, 0, s>>>(GA); break;
    default: gemm_nt<1, 1, 1><<<g, b, 0, s>>>(GA); break;
  }
#undef GA
}

// -------------------- rank-3 correction / attention kernels ----------------

__global__ void k_proj3(const float* __restrict__ src, const float* __restrict__ we,
                        const float* __restrict__ be, float* __restrict__ qe,
                        float* __restrict__ qb) {
  int rh = blockIdx.x;                // (b*N+n)*H + h
  int bi = rh / Hh, h = rh % Hh, t = threadIdx.x;
  float a0 = 0, a1 = 0, a2 = 0, a3 = 0;
  #pragma unroll
  for (int u = 0; u < 2; ++u) {
    int d = t + u * 32;
    float q = src[(long)bi * INNER + h * DHd + d];
    a0 += q * we[0 * INNER + h * DHd + d];
    a1 += q * we[1 * INNER + h * DHd + d];
    a2 += q * we[2 * INNER + h * DHd + d];
    a3 += q * be[h * DHd + d];
  }
  a0 = wred_sum(a0); a1 = wred_sum(a1); a2 = wred_sum(a2); a3 = wred_sum(a3);
  if (t == 0) { qe[rh * 3 + 0] = a0; qe[rh * 3 + 1] = a1; qe[rh * 3 + 2] = a2; qb[rh] = a3; }
}

__global__ void k_softmax(const float* __restrict__ simraw, const float* __restrict__ edges,
                          const float* __restrict__ qe, const float* __restrict__ qb,
                          float* __restrict__ attn, half_t* __restrict__ attnh,
                          half_t* __restrict__ attnT) {
  int row = blockIdx.x;               // (b*H+h)*N + i
  int b = row / (Hh * Nn), h = (row / Nn) % Hh, i = row % Nn, t = threadIdx.x;
  int bih = ((b * Nn + i) * Hh + h) * 3;
  float c0 = qe[bih], c1 = qe[bih + 1], c2 = qe[bih + 2], cb = qb[(b * Nn + i) * Hh + h];
  long bhBase = (long)(b * Hh + h) * Nn * Nn;
  float sv[4];
  #pragma unroll
  for (int u = 0; u < 4; ++u) {
    int j = t + u * 32;
    long eb = (((long)b * Nn + i) * Nn + j) * 3;
    sv[u] = (simraw[(long)row * Nn + j] + cb +
             edges[eb] * c0 + edges[eb + 1] * c1 + edges[eb + 2] * c2) * SCALE;
  }
  float mx = fmaxf(fmaxf(sv[0], sv[1]), fmaxf(sv[2], sv[3]));
  mx = wred_max(mx);
  float sm = 0;
  #pragma unroll
  for (int u = 0; u < 4; ++u) { sv[u] = expf(sv[u] - mx); sm += sv[u]; }
  sm = wred_sum(sm);
  float inv = 1.f / sm;
  #pragma unroll
  for (int u = 0; u < 4; ++u) {
    int j = t + u * 32;
    float a = sv[u] * inv;
    attn[(long)row * Nn + j] = a;
    attnh[(long)row * Nn + j] = (half_t)a;
    attnT[bhBase + (long)j * Nn + i] = (half_t)a;
  }
}

__global__ void k_outcorr(const float* __restrict__ attn, const float* __restrict__ edges,
                          const float* __restrict__ we, const float* __restrict__ be,
                          float* __restrict__ outi, half_t* __restrict__ outih) {
  int rh = blockIdx.x;                // (b*N+i)*H + h
  int bi = rh / Hh, h = rh % Hh, b = bi / Nn, i = bi % Nn, t = threadIdx.x;
  float w0 = 0, w1 = 0, w2 = 0;
  #pragma unroll
  for (int u = 0; u < 4; ++u) {
    int j = t + u * 32;
    float a = attn[(((long)b * Hh + h) * Nn + i) * Nn + j];
    long eb = (((long)b * Nn + i) * Nn + j) * 3;
    w0 += a * edges[eb]; w1 += a * edges[eb + 1]; w2 += a * edges[eb + 2];
  }
  w0 = wred_sum(w0); w1 = wred_sum(w1); w2 = wred_sum(w2);
  #pragma unroll
  for (int u = 0; u < 2; ++u) {
    int d = t + u * 32, hd = h * DHd + d;
    long idx = (long)bi * INNER + hd;
    float o = outi[idx] + w0 * we[hd] + w1 * we[INNER + hd] + w2 * we[2 * INNER + hd] + be[hd];
    outi[idx] = o;
    outih[idx] = (half_t)o;
  }
}

// -------------------------- gating / gelu ----------------------------------

__global__ void k_gr_fwd(const float* __restrict__ x, const float* __restrict__ res,
                         const float* __restrict__ wg, float* __restrict__ out,
                         float* __restrict__ gate) {
  int row = blockIdx.x, t = threadIdx.x;
  float z = 0;
  #pragma unroll
  for (int u = 0; u < 2; ++u) {
    int d = t + u * 32;
    float xv = x[row * Dm + d], rv = res[row * Dm + d];
    z += xv * wg[d] + rv * wg[64 + d] + (xv - rv) * wg[128 + d];
  }
  z = wred_sum(z);
  float g = 1.f / (1.f + expf(-z));
  #pragma unroll
  for (int u = 0; u < 2; ++u) {
    int d = t + u * 32;
    out[row * Dm + d] = x[row * Dm + d] * g + res[row * Dm + d] * (1.f - g);
  }
  if (t == 0) gate[row] = g;
}

__global__ void k_gr_bwd(const float* __restrict__ dout, const float* __restrict__ x,
                         const float* __restrict__ res, const float* __restrict__ wg,
                         const float* __restrict__ gate, float* __restrict__ dx,
                         half_t* __restrict__ dxh, float* __restrict__ dres) {
  int row = blockIdx.x, t = threadIdx.x;
  float g = gate[row];
  float dd = 0;
  #pragma unroll
  for (int u = 0; u < 2; ++u) {
    int d = t + u * 32;
    dd += dout[row * Dm + d] * (x[row * Dm + d] - res[row * Dm + d]);
  }
  dd = wred_sum(dd);
  float dz = dd * g * (1.f - g);
  #pragma unroll
  for (int u = 0; u < 2; ++u) {
    int d = t + u * 32;
    float dv = dout[row * Dm + d];
    float dxv = dv * g + dz * (wg[d] + wg[128 + d]);
    dx[row * Dm + d] = dxv;
    dxh[row * Dm + d] = (half_t)dxv;
    dres[row * Dm + d] = dv * (1.f - g) + dz * (wg[64 + d] - wg[128 + d]);
  }
}

__global__ void k_gelu_fwd(const float* __restrict__ h, half_t* __restrict__ hgh, int n) {
  int i = blockIdx.x * blockDim.x + threadIdx.x;
  for (; i < n; i += gridDim.x * blockDim.x) {
    float v = h[i];
    hgh[i] = (half_t)(v * 0.5f * (1.f + erff(v * 0.70710678118654752f)));
  }
}

__global__ void k_gelu_bwd(const float* __restrict__ h1, const float* __restrict__ dh1g,
                           half_t* __restrict__ dh1h, int n) {
  int i = blockIdx.x * blockDim.x + threadIdx.x;
  for (; i < n; i += gridDim.x * blockDim.x) {
    float v = h1[i];
    float Phi = 0.5f * (1.f + erff(v * 0.70710678118654752f));
    float pdf = 0.3989422804014327f * expf(-0.5f * v * v);
    dh1h[i] = (half_t)(dh1g[i] * (Phi + v * pdf));
  }
}

// ------------------------- backward attention pieces -----------------------

__global__ void k_smax_bwd(const float* __restrict__ dattnG, const float* __restrict__ attn,
                           const float* __restrict__ edges, const float* __restrict__ gve,
                           const float* __restrict__ gbe, float* __restrict__ dsim,
                           half_t* __restrict__ dsimh, half_t* __restrict__ dsimT) {
  int row = blockIdx.x;               // (b*H+h)*N + i
  int b = row / (Hh * Nn), h = (row / Nn) % Hh, i = row % Nn, t = threadIdx.x;
  int bih = ((b * Nn + i) * Hh + h) * 3;
  float c0 = gve[bih], c1 = gve[bih + 1], c2 = gve[bih + 2], cb = gbe[(b * Nn + i) * Hh + h];
  long bhBase = (long)(b * Hh + h) * Nn * Nn;
  float da[4], av[4], dot = 0;
  #pragma unroll
  for (int u = 0; u < 4; ++u) {
    int j = t + u * 32;
    long eb = (((long)b * Nn + i) * Nn + j) * 3;
    da[u] = dattnG[(long)row * Nn + j] + cb +
            edges[eb] * c0 + edges[eb + 1] * c1 + edges[eb + 2] * c2;
    av[u] = attn[(long)row * Nn + j];
    dot += av[u] * da[u];
  }
  dot = wred_sum(dot);
  #pragma unroll
  for (int u = 0; u < 4; ++u) {
    int j = t + u * 32;
    float ds = av[u] * (da[u] - dot);
    dsim[(long)row * Nn + j] = ds;
    dsimh[(long)row * Nn + j] = (half_t)ds;
    dsimT[bhBase + (long)j * Nn + i] = (half_t)ds;
  }
}

__global__ void k_dqcorr(const float* __restrict__ dsim, const float* __restrict__ edges,
                         const float* __restrict__ we, float* __restrict__ dq,
                         half_t* __restrict__ dqh) {
  int rh = blockIdx.x;                // (b*N+i)*H + h
  int bi = rh / Hh, h = rh % Hh, b = bi / Nn, i = bi % Nn, t = threadIdx.x;
  float s0 = 0, s1 = 0, s2 = 0;
  #pragma unroll
  for (int u = 0; u < 4; ++u) {
    int j = t + u * 32;
    float s = dsim[(((long)b * Hh + h) * Nn + i) * Nn + j];
    long eb = (((long)b * Nn + i) * Nn + j) * 3;
    s0 += s * edges[eb]; s1 += s * edges[eb + 1]; s2 += s * edges[eb + 2];
  }
  s0 = wred_sum(s0); s1 = wred_sum(s1); s2 = wred_sum(s2);
  #pragma unroll
  for (int u = 0; u < 2; ++u) {
    int d = t + u * 32, hd = h * DHd + d;
    long idx = (long)bi * INNER + hd;
    float v = dq[idx] + SCALE * (s0 * we[hd] + s1 * we[INNER + hd] + s2 * we[2 * INNER + hd]);
    dq[idx] = v;
    dqh[idx] = (half_t)v;
  }
}

__global__ void k_dedge(const float* __restrict__ dsim, const float* __restrict__ attn,
                        const float* __restrict__ qe, const float* __restrict__ gve,
                        float* __restrict__ dE) {
  int idx = blockIdx.x * blockDim.x + threadIdx.x;     // over B*N*N
  if (idx >= Bsz * Nn * Nn) return;
  int j = idx % Nn, i = (idx / Nn) % Nn, b = idx / (Nn * Nn);
  float a0 = 0, a1 = 0, a2 = 0;
  #pragma unroll
  for (int h = 0; h < Hh; ++h) {
    float s = SCALE * dsim[(((long)b * Hh + h) * Nn + i) * Nn + j];
    float a = attn[(((long)b * Hh + h) * Nn + i) * Nn + j];
    int bih = ((b * Nn + i) * Hh + h) * 3;
    a0 += s * qe[bih]     + a * gve[bih];
    a1 += s * qe[bih + 1] + a * gve[bih + 1];
    a2 += s * qe[bih + 2] + a * gve[bih + 2];
  }
  long eb = (long)idx * 3;
  dE[eb] += a0; dE[eb + 1] += a1; dE[eb + 2] += a2;
}

__global__ void k_grad_init(const float* __restrict__ lw, float* __restrict__ dcur) {
  int i = blockIdx.x * blockDim.x + threadIdx.x;
  if (i < ROWS * Dm) dcur[i] = lw[i % Dm];
}

__global__ void k_dcoords(const float* __restrict__ dE, const int* __restrict__ mask,
                          float* __restrict__ out) {
  int bn = blockIdx.x, b = bn / Nn, n = bn % Nn, t = threadIdx.x;
  float a0 = 0, a1 = 0, a2 = 0;
  #pragma unroll
  for (int u = 0; u < 4; ++u) {
    int q = t + u * 32;
    if (mask[n * Nn + q]) {
      long eb = (((long)b * Nn + n) * Nn + q) * 3;
      a0 += dE[eb]; a1 += dE[eb + 1]; a2 += dE[eb + 2];
    }
    if (mask[q * Nn + n]) {
      long eb = (((long)b * Nn + q) * Nn + n) * 3;
      a0 -= dE[eb]; a1 -= dE[eb + 1]; a2 -= dE[eb + 2];
    }
  }
  a0 = wred_sum(a0); a1 = wred_sum(a1); a2 = wred_sum(a2);
  if (t == 0) { out[bn * 3] = a0; out[bn * 3 + 1] = a1; out[bn * 3 + 2] = a2; }
}

// ------------------------------- host side ---------------------------------

struct LayerP {
  const float *ln1g, *ln1b, *wq, *bq, *wkv, *bkv, *we, *be, *wo, *bo, *wga,
              *ln2g, *ln2b, *w1, *b1, *w2, *b2, *wgf;
};
struct LayerW {   // f16 copies: straight (K,N) for backward, transposed (N,K) for forward
  half_t *wq, *wqT, *wkv, *wkvT, *wo, *woT, *w1, *w1T, *w2, *w2T;
};
struct WS {
  float *stash;
  float *edges, *dE; int* mask;
  float *mean1, *rstd1, *mean2, *rstd2, *gate_a, *gate_f;
  half_t *x_h, *y_h;
  float *q; half_t *q_h, *qT;
  float *kv; half_t *kv_h, *kT, *vT;
  float *qe, *qb, *gve, *gbe;
  float *simraw, *attn; half_t *attn_h, *attnT;
  float *outi; half_t *outi_h;
  float *attn_out, *nodes_mid, *h1, *y2, *tmp_out;
  half_t *h1g_h;
  float *dA, *dB, *dmid, *dy2, *dh1g, *dy, *dattn_out, *dout_inner, *dattnG,
        *dsim, *dq, *dkv, *dx;
  half_t *dy2_h, *dh1_h, *dattn_out_h, *dout_h, *doutT, *dsim_h, *dsimT, *dq_h, *dkv_h;
};

static const long SBH = (long)Nn * Nn;               // per-head attn stride
static const long SHD = (long)DHd * Nn;              // per-head transposed 64x128

static void fwd_layer(hipStream_t s, const LayerP& P, const LayerW& W, WS& w,
                      const float* nin, float* nout) {
  k_ln_fwd<<<ROWS, 32, 0, s>>>(nin, P.ln1g, P.ln1b, w.x_h, w.mean1, w.rstd1);
  gemm_launch(s, 1, 0, 1, w.x_h, W.wqT, P.bq, w.q, w.q_h, ROWS, INNER, Dm,
              Dm, Dm, INNER, 0, 0, 0, 0, 0, 0, 1, 1, 1.f);
  gemm_launch(s, 1, 0, 1, w.x_h, W.wkvT, P.bkv, w.kv, w.kv_h, ROWS, 2 * INNER, Dm,
              Dm, Dm, 2 * INNER, 0, 0, 0, 0, 0, 0, 1, 1, 1.f);
  k_tr_head<<<(Bsz * Hh * DHd * Nn + 255) / 256, 256, 0, s>>>(w.kv, w.vT, 2 * INNER, INNER);
  k_proj3<<<ROWS * Hh, 32, 0, s>>>(w.q, P.we, P.be, w.qe, w.qb);
  // sim = q @ k^T  per (b,h)
  gemm_launch(s, 0, 0, 0, w.q_h, w.kv_h, nullptr, w.simraw, nullptr, Nn, Nn, DHd,
              INNER, 2 * INNER, Nn,
              (long)Nn * INNER, DHd, (long)Nn * 2 * INNER, DHd,
              Hh * SBH, SBH, Hh, Bsz * Hh, 1.f);
  k_softmax<<<Bsz * Hh * Nn, 32, 0, s>>>(w.simraw, w.edges, w.qe, w.qb,
                                         w.attn, w.attn_h, w.attnT);
  // out = attn @ v   (v accessed via per-head transpose vT[d][j])
  gemm_launch(s, 0, 0, 0, w.attn_h, w.vT, nullptr, w.outi, nullptr, Nn, DHd, Nn,
              Nn, Nn, INNER,
              Hh * SBH, SBH, Hh * SHD, SHD,
              (long)Nn * INNER, DHd, Hh, Bsz * Hh, 1.f);
  k_outcorr<<<ROWS * Hh, 32, 0, s>>>(w.attn, w.edges, P.we, P.be, w.outi, w.outi_h);
  gemm_launch(s, 1, 0, 0, w.outi_h, W.woT, P.bo, w.attn_out, nullptr, ROWS, Dm, INNER,
              INNER, INNER, Dm, 0, 0, 0, 0, 0, 0, 1, 1, 1.f);
  k_gr_fwd<<<ROWS, 32, 0, s>>>(w.attn_out, nin, P.wga, w.nodes_mid, w.gate_a);
  k_ln_fwd<<<ROWS, 32, 0, s>>>(w.nodes_mid, P.ln2g, P.ln2b, w.y_h, w.mean2, w.rstd2);
  gemm_launch(s, 1, 0, 0, w.y_h, W.w1T, P.b1, w.h1, nullptr, ROWS, FFD, Dm,
              Dm, Dm, FFD, 0, 0, 0, 0, 0, 0, 1, 1, 1.f);
  k_gelu_fwd<<<256, 256, 0, s>>>(w.h1, w.h1g_h, ROWS * FFD);
  gemm_launch(s, 1, 0, 0, w.h1g_h, W.w2T, P.b2, w.y2, nullptr, ROWS, Dm, FFD,
              FFD, FFD, Dm, 0, 0, 0, 0, 0, 0, 1, 1, 1.f);
  k_gr_fwd<<<ROWS, 32, 0, s>>>(w.y2, w.nodes_mid, P.wgf, nout, w.gate_f);
}

static void bwd_layer(hipStream_t s, const LayerP& P, const LayerW& W, WS& w,
                      const float* nin, const float* dcur, float* dprev) {
  fwd_layer(s, P, W, w, nin, w.tmp_out);   // recompute internals
  int trg = (Bsz * Hh * DHd * Nn + 255) / 256;
  k_tr_head<<<trg, 256, 0, s>>>(w.q, w.qT, INNER, 0);
  k_tr_head<<<trg, 256, 0, s>>>(w.kv, w.kT, 2 * INNER, 0);
  // ---- FFN branch ----
  k_gr_bwd<<<ROWS, 32, 0, s>>>(dcur, w.y2, w.nodes_mid, P.wgf, w.gate_f,
                               w.dy2, w.dy2_h, w.dmid);
  gemm_launch(s, 0, 0, 0, w.dy2_h, W.w2, nullptr, w.dh1g, nullptr, ROWS, FFD, Dm,
              Dm, Dm, FFD, 0, 0, 0, 0, 0, 0, 1, 1, 1.f);
  k_gelu_bwd<<<256, 256, 0, s>>>(w.h1, w.dh1g, w.dh1_h, ROWS * FFD);
  gemm_launch(s, 0, 0, 0, w.dh1_h, W.w1, nullptr, w.dy, nullptr, ROWS, Dm, FFD,
              FFD, FFD, Dm, 0, 0, 0, 0, 0, 0, 1, 1, 1.f);
  k_ln_bwd<<<ROWS, 32, 0, s>>>(w.dy, w.nodes_mid, P.ln2g, w.mean2, w.rstd2, w.dmid, 1);
  // ---- attention branch ----
  k_gr_bwd<<<ROWS, 32, 0, s>>>(w.dmid, w.attn_out, nin, P.wga, w.gate_a,
                               w.dattn_out, w.dattn_out_h, dprev);
  gemm_launch(s, 0, 0, 1, w.dattn_out_h, W.wo, nullptr, w.dout_inner, w.dout_h,
              ROWS, INNER, Dm, Dm, Dm, INNER, 0, 0, 0, 0, 0, 0, 1, 1, 1.f);
  k_tr_head<<<trg, 256, 0, s>>>(w.dout_inner, w.doutT, INNER, 0);
  k_proj3<<<ROWS * Hh, 32, 0, s>>>(w.dout_inner, P.we, P.be, w.gve, w.gbe);
  // dattnG = dout @ v^T
  gemm_launch(s, 0, 0, 0, w.dout_h, w.kv_h + INNER, nullptr, w.dattnG, nullptr,
              Nn, Nn, DHd, INNER, 2 * INNER, Nn,
              (long)Nn * INNER, DHd, (long)Nn * 2 * INNER, DHd,
              Hh * SBH, SBH, Hh, Bsz * Hh, 1.f);
  k_smax_bwd<<<Bsz * Hh * Nn, 32, 0, s>>>(w.dattnG, w.attn, w.edges, w.gve, w.gbe,
                                          w.dsim, w.dsim_h, w.dsimT);
  k_dedge<<<(Bsz * Nn * Nn + 255) / 256, 256, 0, s>>>(w.dsim, w.attn, w.qe, w.gve, w.dE);
  // dq = SCALE * dsim @ k   (k via kT)
  gemm_launch(s, 0, 0, 0, w.dsim_h, w.kT, nullptr, w.dq, nullptr, Nn, DHd, Nn,
              Nn, Nn, INNER,
              Hh * SBH, SBH, Hh * SHD, SHD,
              (long)Nn * INNER, DHd, Hh, Bsz * Hh, SCALE);
  k_dqcorr<<<ROWS * Hh, 32, 0, s>>>(w.dsim, w.edges, P.we, w.dq, w.dq_h);
  // dk = SCALE * dsim^T @ q  (dsimT row-major, q via qT)
  gemm_launch(s, 0, 0, 1, w.dsimT, w.qT, nullptr, w.dkv, w.dkv_h, Nn, DHd, Nn,
              Nn, Nn, 2 * INNER,
              Hh * SBH, SBH, Hh * SHD, SHD,
              (long)Nn * 2 * INNER, DHd, Hh, Bsz * Hh, SCALE);
  // dv = attn^T @ dout  (attnT row-major, dout via doutT)
  gemm_launch(s, 0, 0, 1, w.attnT, w.doutT, nullptr, w.dkv + INNER, w.dkv_h + INNER,
              Nn, DHd, Nn, Nn, Nn, 2 * INNER,
              Hh * SBH, SBH, Hh * SHD, SHD,
              (long)Nn * 2 * INNER, DHd, Hh, Bsz * Hh, 1.f);
  // dx = dq @ wq^T + dkv @ wkv^T
  gemm_launch(s, 0, 0, 0, w.dq_h, W.wq, nullptr, w.dx, nullptr, ROWS, Dm, INNER,
              INNER, INNER, Dm, 0, 0, 0, 0, 0, 0, 1, 1, 1.f);
  gemm_launch(s, 0, 1, 0, w.dkv_h, W.wkv, nullptr, w.dx, nullptr, ROWS, Dm, 2 * INNER,
              2 * INNER, 2 * INNER, Dm, 0, 0, 0, 0, 0, 0, 1, 1, 1.f);
  k_ln_bwd<<<ROWS, 32, 0, s>>>(w.dx, nin, P.ln1g, w.mean1, w.rstd1, dprev, 1);
}

extern "C" void kernel_launch(void* const* d_in, const int* in_sizes, int n_in,
                              void* d_out, int out_size, void* d_ws, size_t ws_size,
                              hipStream_t stream) {
  (void)in_sizes; (void)n_in; (void)out_size; (void)ws_size;
  const int*   indices  = (const int*)d_in[0];
  const float* coords   = (const float*)d_in[1];
  const int*   bonds    = (const int*)d_in[2];
  const float* noise    = (const float*)d_in[3];
  const float* emb      = (const float*)d_in[4];
  const float* linear_w = (const float*)d_in[5];
  LayerP P[NLAY];
  for (int l = 0; l < NLAY; ++l) {
    const float** f = (const float**)&P[l];
    for (int k = 0; k < 18; ++k) f[k] = (const float*)d_in[7 + l * 18 + k];
  }

  char* base = (char*)d_ws; size_t off = 0;
  auto alloc = [&](size_t bytes) { void* p = base + off; off = (off + bytes + 255) & ~(size_t)255; return p; };
  auto F = [&](size_t n) { return (float*)alloc(n * 4); };
  auto Hp = [&](size_t n) { return (half_t*)alloc(n * 2); };

  WS w{};
  w.stash  = F((size_t)13 * ROWS * Dm);
  w.edges  = F((size_t)Bsz * Nn * Nn * 3);
  w.dE     = F((size_t)Bsz * Nn * Nn * 3);
  w.mask   = (int*)alloc((size_t)Nn * Nn * 4);
  w.mean1 = F(ROWS); w.rstd1 = F(ROWS); w.mean2 = F(ROWS); w.rstd2 = F(ROWS);
  w.gate_a = F(ROWS); w.gate_f = F(ROWS);
  w.x_h = Hp(ROWS * Dm); w.y_h = Hp(ROWS * Dm);
  w.q = F((size_t)ROWS * INNER); w.q_h = Hp((size_t)ROWS * INNER);
  w.qT = Hp((size_t)Bsz * Hh * SHD);
  w.kv = F((size_t)ROWS * 2 * INNER); w.kv_h = Hp((size_t)ROWS * 2 * INNER);
  w.kT = Hp((size_t)Bsz * Hh * SHD); w.vT = Hp((size_t)Bsz * Hh * SHD);
  w.qe = F(ROWS * Hh * 3); w.qb = F(ROWS * Hh);
  w.gve = F(ROWS * Hh * 3); w.gbe = F(ROWS * Hh);
  w.simraw = F((size_t)Bsz * Hh * Nn * Nn);
  w.attn = F((size_t)Bsz * Hh * Nn * Nn);
  w.attn_h = Hp((size_t)Bsz * Hh * Nn * Nn); w.attnT = Hp((size_t)Bsz * Hh * Nn * Nn);
  w.outi = F((size_t)ROWS * INNER); w.outi_h = Hp((size_t)ROWS * INNER);
  w.attn_out = F(ROWS * Dm); w.nodes_mid = F(ROWS * Dm);
  w.h1 = F((size_t)ROWS * FFD); w.h1g_h = Hp((size_t)ROWS * FFD);
  w.y2 = F(ROWS * Dm); w.tmp_out = F(ROWS * Dm);
  w.dA = F(ROWS * Dm); w.dB = F(ROWS * Dm); w.dmid = F(ROWS * Dm);
  w.dy2 = F(ROWS * Dm); w.dy2_h = Hp(ROWS * Dm);
  w.dh1g = F((size_t)ROWS * FFD); w.dh1_h = Hp((size_t)ROWS * FFD);
  w.dy = F(ROWS * Dm);
  w.dattn_out = F(ROWS * Dm); w.dattn_out_h = Hp(ROWS * Dm);
  w.dout_inner = F((size_t)ROWS * INNER); w.dout_h = Hp((size_t)ROWS * INNER);
  w.doutT = Hp((size_t)Bsz * Hh * SHD);
  w.dattnG = F((size_t)Bsz * Hh * Nn * Nn);
  w.dsim = F((size_t)Bsz * Hh * Nn * Nn);
  w.dsim_h = Hp((size_t)Bsz * Hh * Nn * Nn); w.dsimT = Hp((size_t)Bsz * Hh * Nn * Nn);
  w.dq = F((size_t)ROWS * INNER); w.dq_h = Hp((size_t)ROWS * INNER);
  w.dkv = F((size_t)ROWS * 2 * INNER); w.dkv_h = Hp((size_t)ROWS * 2 * INNER);
  w.dx = F(ROWS * Dm);

  LayerW W[NLAY];
  for (int l = 0; l < NLAY; ++l) {
    W[l].wq  = Hp(Dm * INNER);      W[l].wqT  = Hp(Dm * INNER);
    W[l].wkv = Hp(Dm * 2 * INNER);  W[l].wkvT = Hp(Dm * 2 * INNER);
    W[l].wo  = Hp(INNER * Dm);      W[l].woT  = Hp(INNER * Dm);
    W[l].w1  = Hp(Dm * FFD);        W[l].w1T  = Hp(Dm * FFD);
    W[l].w2  = Hp(FFD * Dm);        W[l].w2T  = Hp(FFD * Dm);
  }

  // ---- setup ----
  k_zero<<<256, 256, 0, stream>>>(w.edges, Bsz * Nn * Nn * 3);
  k_zero<<<256, 256, 0, stream>>>(w.dE, Bsz * Nn * Nn * 3);
  k_zero<<<64, 256, 0, stream>>>((float*)w.mask, Nn * Nn);
  for (int l = 0; l < NLAY; ++l) {
    k_cvt<<<64, 256, 0, stream>>>(P[l].wq,  W[l].wq,  Dm * INNER);
    k_cvt_T<<<64, 256, 0, stream>>>(P[l].wq, W[l].wqT, Dm, INNER);
    k_cvt<<<128, 256, 0, stream>>>(P[l].wkv, W[l].wkv, Dm * 2 * INNER);
    k_cvt_T<<<128, 256, 0, stream>>>(P[l].wkv, W[l].wkvT, Dm, 2 * INNER);
    k_cvt<<<64, 256, 0, stream>>>(P[l].wo,  W[l].wo,  INNER * Dm);
    k_cvt_T<<<64, 256, 0, stream>>>(P[l].wo, W[l].woT, INNER, Dm);
    k_cvt<<<32, 256, 0, stream>>>(P[l].w1,  W[l].w1,  Dm * FFD);
    k_cvt_T<<<32, 256, 0, stream>>>(P[l].w1, W[l].w1T, Dm, FFD);
    k_cvt<<<32, 256, 0, stream>>>(P[l].w2,  W[l].w2,  FFD * Dm);
    k_cvt_T<<<32, 256, 0, stream>>>(P[l].w2, W[l].w2T, FFD, Dm);
  }
  k_build_nodes<<<(ROWS * Dm + 255) / 256, 256, 0, stream>>>(indices, emb, noise, w.stash);
  k_build_edges<<<1, En, 0, stream>>>(bonds, coords, w.edges, w.mask);

  // ---- forward: 12 applications, checkpoint every layer input ----
  for (int a = 0; a < APPS; ++a) {
    int l = a % NLAY;
    fwd_layer(stream, P[l], W[l], w,
              w.stash + (size_t)a * ROWS * Dm,
              w.stash + (size_t)(a + 1) * ROWS * Dm);
  }

  // ---- backward ----
  k_grad_init<<<(ROWS * Dm + 255) / 256, 256, 0, stream>>>(linear_w, w.dA);
  float* dcur = w.dA; float* dprev = w.dB;
  for (int a = APPS - 1; a >= 0; --a) {
    int l = a % NLAY;
    bwd_layer(stream, P[l], W[l], w, w.stash + (size_t)a * ROWS * Dm, dcur, dprev);
    float* t = dcur; dcur = dprev; dprev = t;
  }

  k_dcoords<<<ROWS, 32, 0, stream>>>(w.dE, w.mask, (float*)d_out);
}
